// PIXBAEncoder_11682311045899
// MI455X (gfx1250) — compile-verified, matrix operand order checked by reading, not run on hardware
//
#include <hip/hip_runtime.h>
#include <hip/hip_bf16.h>
#include <math.h>
#include <cstdint>

// ---------------------------------------------------------------------------
// Mamba (4 blocks) forward for MI455X / gfx1250, wave32 + WMMA + TDM.
//   B=2, L=2048, D=1024, d_inner=2048, N_state=16, dt_rank=64, d_conv=4
// GEMMs: NT WMMA (f32 -> f16 operands, f32 accumulate), float4-vectorized
//        LDS staging, BK=64 (8 v_wmma per barrier pair per wave).
// Scan:  latency-bound (2048 serial steps) -> TDM double-buffers 8-timestep
//        tiles of u/dt/z into LDS (tensor_load_to_lds + s_wait_tensorcnt).
// ---------------------------------------------------------------------------

typedef __attribute__((ext_vector_type(16))) _Float16 v16h;
typedef __attribute__((ext_vector_type(8)))  float    v8f;
typedef __attribute__((ext_vector_type(4)))  _Float16 h4;

#define BM 128
#define BN 64
#define BK 64

// ---------------------------------------------------------------------------
// Tensor Data Mover helper (gfx1250). D# built per CDNA5 ISA §8.3/8.4.
// ---------------------------------------------------------------------------
#if defined(__has_builtin)
#if __has_builtin(__builtin_amdgcn_tensor_load_to_lds)
#define HAVE_TDM 1
#endif
#endif

#ifdef HAVE_TDM
typedef __attribute__((ext_vector_type(4))) unsigned int u32x4;
typedef __attribute__((ext_vector_type(4))) int          i32x4v;
typedef __attribute__((ext_vector_type(8))) int          i32x8v;

// 2-D tile load: tileX contiguous f32 elements per row, tileY rows,
// row stride = strideElems f32 elements, packed row-major into LDS.
__device__ __forceinline__ void tdm_load_tile_2d(void* ldsDst, const float* gsrc,
                                                 int tileX, int tileY,
                                                 int strideElems) {
  uint64_t ga  = (uint64_t)(uintptr_t)gsrc;
  uint32_t lds = (uint32_t)(uintptr_t)ldsDst;
  u32x4 g0;
  g0[0] = 1u;                                           // count=1, user mode
  g0[1] = lds;                                          // lds_addr
  g0[2] = (uint32_t)ga;                                 // global_addr[31:0]
  g0[3] = ((uint32_t)(ga >> 32) & 0x01FFFFFFu)          // global_addr[56:32]
          | (2u << 30);                                 // type=2 ("image")
  i32x8v g1;
  g1[0] = (int)(2u << 16);                              // data_size = 4 bytes
  g1[1] = (int)(((uint32_t)tileX & 0xFFFFu) << 16);     // tensor_dim0[15:0]
  g1[2] = (int)(((uint32_t)tileY & 0xFFFFu) << 16);     // dim0 hi | tensor_dim1 lo
  g1[3] = (int)(((uint32_t)tileX & 0xFFFFu) << 16);     // dim1 hi | tile_dim0
  g1[4] = (int)((uint32_t)tileY & 0xFFFFu);             // tile_dim1 (tile_dim2=0)
  g1[5] = strideElems;                                  // tensor_dim0_stride lo32
  g1[6] = 0;                                            // stride hi | dim1_stride lo
  g1[7] = 0;
  i32x4v gz = {0, 0, 0, 0};
#if __clang_major__ >= 23
  i32x8v gz8 = {0, 0, 0, 0, 0, 0, 0, 0};
  __builtin_amdgcn_tensor_load_to_lds(g0, g1, gz, gz, gz8, 0);
#else
  __builtin_amdgcn_tensor_load_to_lds(g0, g1, gz, gz, 0);
#endif
}
#endif  // HAVE_TDM

// ---------------------------------------------------------------------------
// C[M,N] = act( A[M,K] (row-major, stride lda) * W[N,K]^T + bias )
// act: 0 = none, 1 = softplus(x + bias[col])
// ---------------------------------------------------------------------------
__global__ __launch_bounds__(256)
void gemm_nt_wmma(const float* __restrict__ A, const float* __restrict__ W,
                  const float* __restrict__ bias, float* __restrict__ C,
                  int M, int N, int K, int lda, int act) {
  __shared__ __attribute__((aligned(32))) _Float16 Ash[BM * BK];  // 16 KB
  __shared__ __attribute__((aligned(32))) _Float16 Wsh[BN * BK];  //  8 KB

  const int tid  = threadIdx.x;
  const int wave = tid >> 5;   // 8 waves of 32 (wave32)
  const int lane = tid & 31;
  const int brow = blockIdx.x * BM;
  const int bcol = blockIdx.y * BN;

  v8f acc[4];
#pragma unroll
  for (int j = 0; j < 4; ++j)
    acc[j] = (v8f){0.f, 0.f, 0.f, 0.f, 0.f, 0.f, 0.f, 0.f};

  for (int k0 = 0; k0 < K; k0 += BK) {
    // ---- batched float4 global loads (one wait, not one per element) ----
    float4 ar[(BM * BK / 4) / 256];            // 8 per thread
#pragma unroll
    for (int i = 0; i < (BM * BK / 4) / 256; ++i) {
      int q = tid + 256 * i;                   // quad index
      int r = q >> 4;                          // BK/4 = 16 quads per row
      int cq = q & 15;
      ar[i] = *(const float4*)&A[(size_t)(brow + r) * lda + (k0 + cq * 4)];
    }
    float4 wr[(BN * BK / 4) / 256];            // 4 per thread
#pragma unroll
    for (int i = 0; i < (BN * BK / 4) / 256; ++i) {
      int q = tid + 256 * i;
      int r = q >> 4;
      int cq = q & 15;
      int n = bcol + r;
      wr[i] = (n < N) ? *(const float4*)&W[(size_t)n * K + (k0 + cq * 4)]
                      : make_float4(0.f, 0.f, 0.f, 0.f);
    }
    // ---- convert f32 -> f16 and store to LDS as 8-byte packets ----
#pragma unroll
    for (int i = 0; i < (BM * BK / 4) / 256; ++i) {
      int q = tid + 256 * i;
      int r = q >> 4, cq = q & 15;
      h4 hv = {(_Float16)ar[i].x, (_Float16)ar[i].y,
               (_Float16)ar[i].z, (_Float16)ar[i].w};
      *(h4*)&Ash[r * BK + cq * 4] = hv;
    }
#pragma unroll
    for (int i = 0; i < (BN * BK / 4) / 256; ++i) {
      int q = tid + 256 * i;
      int r = q >> 4, cq = q & 15;
      h4 hv = {(_Float16)wr[i].x, (_Float16)wr[i].y,
               (_Float16)wr[i].z, (_Float16)wr[i].w};
      *(h4*)&Wsh[r * BK + cq * 4] = hv;
    }
    __syncthreads();

    if (k0 + BK < K) {
      // gfx1250 global_prefetch_b8 on the next A tile
      __builtin_prefetch(&A[(size_t)(brow + (tid >> 1)) * lda + (k0 + BK)], 0, 1);
    }

    const int r16 = lane & 15;
    const int kh  = lane >> 4;
#pragma unroll
    for (int kk = 0; kk < 2; ++kk) {           // two 16x16x32 steps per BK=64
      v16h a = *(const v16h*)&Ash[(wave * 16 + r16) * BK + kk * 32 + kh * 16];
#pragma unroll
      for (int j = 0; j < 4; ++j) {
        v16h b = *(const v16h*)&Wsh[(j * 16 + r16) * BK + kk * 32 + kh * 16];
        acc[j] = __builtin_amdgcn_wmma_f32_16x16x32_f16(
            false, a, false, b, (short)0, acc[j], false, false);
      }
    }
    __syncthreads();
  }

  // D layout: VGPR r, lanes 0-15 -> (M=r, N=lane); lanes 16-31 -> (M=r+8, N=lane-16)
  const int col0  = lane & 15;
  const int rbase = brow + wave * 16 + ((lane >> 4) << 3);
#pragma unroll
  for (int j = 0; j < 4; ++j) {
    int col = bcol + j * 16 + col0;
    if (col < N) {
#pragma unroll
      for (int r = 0; r < 8; ++r) {
        float v = acc[j][r];
        if (act == 1) {
          v += bias[col];
          v = (v > 20.f) ? v : log1pf(__expf(v));   // softplus
        }
        C[(size_t)(rbase + r) * N + col] = v;
      }
    }
  }
}

// ---------------------------------------------------------------------------
// Depthwise causal conv(4) + SiLU over xi = xz[..., :d_inner].
// ---------------------------------------------------------------------------
__global__ __launch_bounds__(256)
void conv_silu(const float* __restrict__ xz, const float* __restrict__ cw,
               const float* __restrict__ cb, float* __restrict__ u) {
  const int DI = 2048, L = 2048;
  size_t g = (size_t)blockIdx.x * 256 + threadIdx.x;   // over B*L*DI
  int d = (int)(g % DI);
  size_t bt = g / DI;
  int t = (int)(bt % L);
  int b = (int)(bt / L);
  float acc = cb[d];
#pragma unroll
  for (int j = 0; j < 4; ++j) {
    int tt = t - 3 + j;
    if (tt >= 0)
      acc += xz[((size_t)b * L + tt) * 4096 + d] * cw[d * 4 + j];
  }
  u[g] = acc / (1.f + __expf(-acc));                   // SiLU
}

// ---------------------------------------------------------------------------
// Selective scan, fused with u*D skip and SiLU(z) gate.
// One thread per (b, d); blocks map to a single batch index b.
// TDM double-buffers 8-timestep tiles of u/dt/z into LDS.
// ---------------------------------------------------------------------------
#define TQ 8   // timesteps per LDS tile

__global__ __launch_bounds__(256)
void scan_fused(const float* __restrict__ u, const float* __restrict__ dt,
                const float* __restrict__ dbc, const float* __restrict__ xz,
                const float* __restrict__ A_log, const float* __restrict__ Dp,
                float* __restrict__ y) {
  const int DI = 2048, L = 2048, NS = 16;
  __shared__ __attribute__((aligned(16))) float Ubuf[2][TQ * 256];   // 16 KB
  __shared__ __attribute__((aligned(16))) float Tbuf[2][TQ * 256];   // 16 KB
  __shared__ __attribute__((aligned(16))) float Zbuf[2][TQ * 256];   // 16 KB
  __shared__ float BCbuf[2][TQ][32];                                 //  2 KB

  const int tid = threadIdx.x;
  const int g   = blockIdx.x * 256 + tid;   // 0 .. B*DI-1
  const int d   = g % DI;
  const int b   = g / DI;                   // uniform per block
  const int d0  = (blockIdx.x * 256) % DI;

  float A[NS], h[NS];
#pragma unroll
  for (int n = 0; n < NS; ++n) {
    A[n] = -__expf(A_log[d * NS + n]);
    h[n] = 0.f;
  }
  const float Dd = Dp[d];
  const int NT = L / TQ;                    // 256 tiles

#ifdef HAVE_TDM
  if (tid == 0) {                           // wave0 issues TDM (EXEC ignored)
    tdm_load_tile_2d(&Ubuf[0][0], &u [(size_t)b * L * DI + d0],        256, TQ, DI);
    tdm_load_tile_2d(&Tbuf[0][0], &dt[(size_t)b * L * DI + d0],        256, TQ, DI);
    tdm_load_tile_2d(&Zbuf[0][0], &xz[(size_t)b * L * 4096 + DI + d0], 256, TQ, 4096);
  }
#else
#pragma unroll
  for (int s = 0; s < TQ; ++s) {
    size_t row = (size_t)b * L + s;
    Ubuf[0][s * 256 + tid] = u [row * DI + d0 + tid];
    Tbuf[0][s * 256 + tid] = dt[row * DI + d0 + tid];
    Zbuf[0][s * 256 + tid] = xz[row * 4096 + DI + d0 + tid];
  }
#endif

  for (int tile = 0; tile < NT; ++tile) {
    const int cur = tile & 1;
    const int t0  = tile * TQ;

#ifdef HAVE_TDM
    if (tid == 0) {
      if (tile + 1 < NT) {
        size_t row = (size_t)b * L + (t0 + TQ);
        tdm_load_tile_2d(&Ubuf[cur ^ 1][0], &u [row * DI + d0],        256, TQ, DI);
        tdm_load_tile_2d(&Tbuf[cur ^ 1][0], &dt[row * DI + d0],        256, TQ, DI);
        tdm_load_tile_2d(&Zbuf[cur ^ 1][0], &xz[row * 4096 + DI + d0], 256, TQ, 4096);
        __builtin_amdgcn_s_wait_tensorcnt((short)3);   // current tile done
      } else {
        __builtin_amdgcn_s_wait_tensorcnt((short)0);
      }
    }
#else
    if (tile + 1 < NT) {
#pragma unroll
      for (int s = 0; s < TQ; ++s) {
        size_t row = (size_t)b * L + (t0 + TQ + s);
        Ubuf[cur ^ 1][s * 256 + tid] = u [row * DI + d0 + tid];
        Tbuf[cur ^ 1][s * 256 + tid] = dt[row * DI + d0 + tid];
        Zbuf[cur ^ 1][s * 256 + tid] = xz[row * 4096 + DI + d0 + tid];
      }
    }
#endif
    // B/C vectors for this tile: 256 threads load exactly TQ*32 values.
    {
      int tt = tid >> 5, n = tid & 31;
      BCbuf[cur][tt][n] = dbc[((size_t)b * L + t0 + tt) * 96 + 64 + n];
    }
    __syncthreads();

#pragma unroll
    for (int s = 0; s < TQ; ++s) {
      float ut  = Ubuf[cur][s * 256 + tid];
      float dtt = Tbuf[cur][s * 256 + tid];
      float zz  = Zbuf[cur][s * 256 + tid];
      float du  = dtt * ut;
      float yv  = 0.f;
#pragma unroll
      for (int n = 0; n < NS; ++n) {
        float dA = __expf(dtt * A[n]);
        h[n] = dA * h[n] + du * BCbuf[cur][s][n];
        yv  += h[n] * BCbuf[cur][s][16 + n];
      }
      float gate = zz / (1.f + __expf(-zz));
      y[((size_t)b * L + t0 + s) * DI + d] = (yv + ut * Dd) * gate;
    }
    __syncthreads();
  }
}

// ---------------------------------------------------------------------------
extern "C" void kernel_launch(void* const* d_in, const int* in_sizes, int n_in,
                              void* d_out, int out_size, void* d_ws, size_t ws_size,
                              hipStream_t stream) {
  (void)in_sizes; (void)n_in; (void)out_size; (void)ws_size;
  const float* x0     = (const float*)d_in[0];  // (2, 2048, 1024)
  const float* in_w   = (const float*)d_in[1];  // (4, 4096, 1024)
  const float* conv_w = (const float*)d_in[2];  // (4, 2048, 1, 4)
  const float* conv_b = (const float*)d_in[3];  // (4, 2048)
  const float* xp_w   = (const float*)d_in[4];  // (4, 96, 2048)
  const float* dtp_w  = (const float*)d_in[5];  // (4, 2048, 64)
  const float* dtp_b  = (const float*)d_in[6];  // (4, 2048)
  const float* A_log  = (const float*)d_in[7];  // (4, 2048, 16)
  const float* D_par  = (const float*)d_in[8];  // (4, 2048)
  const float* out_w  = (const float*)d_in[9];  // (4, 1024, 2048)
  float* out = (float*)d_out;

  const int BATCH = 2, L = 2048, D = 1024, DI = 2048;
  const int M = BATCH * L;                       // 4096 flattened rows

  // Workspace carve-up (floats). Total ~194 MiB.
  float* ws  = (float*)d_ws;
  float* xz  = ws;                               // M * 4096
  float* u   = xz  + (size_t)M * 2 * DI;         // M * DI
  float* dbc = u   + (size_t)M * DI;             // M * 96
  float* dt  = dbc + (size_t)M * 96;             // M * DI
  float* y   = dt  + (size_t)M * DI;             // M * DI
  float* xa  = y   + (size_t)M * DI;             // M * D
  float* xb  = xa  + (size_t)M * D;              // M * D

  const float* xcur = x0;
  for (int i = 0; i < 4; ++i) {
    const float* iw  = in_w   + (size_t)i * 2 * DI * D;
    const float* cwi = conv_w + (size_t)i * DI * 4;
    const float* cbi = conv_b + (size_t)i * DI;
    const float* xpi = xp_w   + (size_t)i * 96 * DI;
    const float* dwi = dtp_w  + (size_t)i * DI * 64;
    const float* dbi = dtp_b  + (size_t)i * DI;
    const float* Ali = A_log  + (size_t)i * DI * 16;
    const float* Dpi = D_par  + (size_t)i * DI;
    const float* owi = out_w  + (size_t)i * D * DI;

    // 1) in_proj: xz = x @ in_w^T          [M x 4096], K=1024
    {
      dim3 grid(M / BM, (2 * DI) / BN);
      gemm_nt_wmma<<<grid, 256, 0, stream>>>(xcur, iw, nullptr, xz,
                                             M, 2 * DI, D, D, 0);
    }
    // 2) depthwise conv + SiLU -> u        [M x 2048]
    conv_silu<<<((size_t)M * DI) / 256, 256, 0, stream>>>(xz, cwi, cbi, u);

    // 3) x_proj: dbc = u @ x_proj_w^T      [M x 96], K=2048 (tail tile guarded)
    {
      dim3 grid(M / BM, (96 + BN - 1) / BN);
      gemm_nt_wmma<<<grid, 256, 0, stream>>>(u, xpi, nullptr, dbc,
                                             M, 96, DI, DI, 0);
    }
    // 4) dt_proj + softplus: dt = softplus(dbc[:, :64] @ dt_w^T + b)
    //    [M x 2048], K=64, A row stride = 96
    {
      dim3 grid(M / BM, DI / BN);
      gemm_nt_wmma<<<grid, 256, 0, stream>>>(dbc, dwi, dbi, dt,
                                             M, DI, 64, 96, 1);
    }
    // 5) selective scan (fused u*D skip + SiLU(z) gate) -> y  [M x 2048]
    scan_fused<<<(BATCH * DI) / 256, 256, 0, stream>>>(u, dt, dbc, xz,
                                                       Ali, Dpi, y);
    // 6) out_proj: x_next = y @ out_w^T    [M x 1024], K=2048
    float* dest = (i == 3) ? out : ((i & 1) ? xb : xa);
    {
      dim3 grid(M / BM, D / BN);
      gemm_nt_wmma<<<grid, 256, 0, stream>>>(y, owi, nullptr, dest,
                                             M, D, DI, DI, 0);
    }
    xcur = dest;
  }
}
